// _LlamaAttention_1219770712166
// MI455X (gfx1250) — matrix operand order repaired and checked by reading, not compiled, not run on hardware
//
#include <hip/hip_runtime.h>
#include <hip/hip_bf16.h>

// ---------------------------------------------------------------------------
// Llama attention prefill for gfx1250 (MI455X), bf16 WMMA compute path.
// All device buffers < 2 GiB -> 32-bit element offsets everywhere so the
// backend can use saddr + 32-bit voffset addressing instead of u64 chains.
// ---------------------------------------------------------------------------

typedef __bf16 bf16;
typedef __attribute__((ext_vector_type(16))) __bf16 v16bf;
typedef __attribute__((ext_vector_type(8)))  __bf16 v8bf;
typedef __attribute__((ext_vector_type(8)))  float  v8f;
typedef __attribute__((ext_vector_type(4)))  float  v4f;

#define T_SEQ 2048
#define HID   4096
#define NH    32
#define NKV   8
#define HD    128

__device__ __forceinline__ v8f wmma_bf16(v16bf a, v16bf b, v8f c) {
  // D(16x16 f32) = A(16x32 bf16) x B(32x16 bf16) + C
  return __builtin_amdgcn_wmma_f32_16x16x32_bf16(
      /*neg_a=*/false, a, /*neg_b=*/false, b,
      /*c_mod=*/(short)0, c, /*reuse_a=*/false, /*reuse_b=*/false);
}

union Frag16 { v16bf v; v8bf h[2]; };

// A-fragment (16x32 MxK): lane l<16 -> row m0+l, K kk..kk+15;
// lane>=16 -> row m0+(l-16), K kk+16..kk+31. off precomputed.
__device__ __forceinline__ v16bf load_a_frag(const bf16* __restrict__ base,
                                             unsigned off) {
  Frag16 u;
  u.h[0] = *(const v8bf*)(base + off);
  u.h[1] = *(const v8bf*)(base + off + 8);
  return u.v;
}

// B-fragment (32x16 KxN), column n = K-contiguous row n of source:
// lane l<16 -> col n0+l, K kk..kk+15 ; lane>=16 -> col n0+(l-16), K kk+16..+31
__device__ __forceinline__ v16bf load_b_frag_bf16(const bf16* __restrict__ base,
                                                  unsigned off) {
  Frag16 u;
  u.h[0] = *(const v8bf*)(base + off);
  u.h[1] = *(const v8bf*)(base + off + 8);
  return u.v;
}

// Same but fp32 source rows (weights), converted on the fly (v_cvt_pk_bf16).
__device__ __forceinline__ v16bf load_b_frag_f32(const float* __restrict__ base,
                                                 unsigned off) {
  v4f f0 = *(const v4f*)(base + off);
  v4f f1 = *(const v4f*)(base + off + 4);
  v4f f2 = *(const v4f*)(base + off + 8);
  v4f f3 = *(const v4f*)(base + off + 12);
  v16bf t;
#pragma unroll
  for (int e = 0; e < 4; ++e) {
    t[e]      = (bf16)f0[e];
    t[4 + e]  = (bf16)f1[e];
    t[8 + e]  = (bf16)f2[e];
    t[12 + e] = (bf16)f3[e];
  }
  return t;
}

__device__ __forceinline__ v8f zero8() {
  v8f z;
#pragma unroll
  for (int e = 0; e < 8; ++e) z[e] = 0.f;
  return z;
}

// ---------------------------------------------------------------------------
// 1) fp32 -> bf16 elementwise
// ---------------------------------------------------------------------------
__global__ void k_cvt_bf16(const float* __restrict__ in, bf16* __restrict__ out,
                           int n) {
  int i = blockIdx.x * blockDim.x + threadIdx.x;
  if (i < n) out[i] = (bf16)in[i];
}

// ---------------------------------------------------------------------------
// 2) GEMM: C[M,N] (fp32) = A_bf16[M,4096] @ W_f32[N,4096]^T
//    Block 256 threads (8 waves), block tile 128x128, wave tile 32x64
//    -> 8 WMMAs per K-step of 32.
// ---------------------------------------------------------------------------
__global__ __launch_bounds__(256)
void k_gemm_bf16w32(const bf16* __restrict__ A, const float* __restrict__ W,
                    float* __restrict__ C, int ldc) {
  const int lane = threadIdx.x & 31;
  const int wave = threadIdx.x >> 5;
  const int l16  = lane & 15;
  const int hl   = lane >> 4;
  const int m0   = blockIdx.y * 128 + (wave >> 1) * 32;   // 4 waves along M
  const int n0   = blockIdx.x * 128 + (wave & 1) * 64;    // 2 waves along N

  v8f c[2][4];
#pragma unroll
  for (int i = 0; i < 2; ++i)
#pragma unroll
    for (int j = 0; j < 4; ++j) c[i][j] = zero8();

  // 32-bit base offsets (K = HID = 4096 -> shifts, not muls)
  unsigned aOff = (unsigned)(m0 + l16) * HID + (hl ? 16u : 0u);
  unsigned bOff = (unsigned)(n0 + l16) * HID + (hl ? 16u : 0u);

  for (int kk = 0; kk < HID; kk += 32) {
    v16bf a[2], b[4];
#pragma unroll
    for (int i = 0; i < 2; ++i)
      a[i] = load_a_frag(A, aOff + (unsigned)(16 * i) * HID + kk);
#pragma unroll
    for (int j = 0; j < 4; ++j)
      b[j] = load_b_frag_f32(W, bOff + (unsigned)(16 * j) * HID + kk);
#pragma unroll
    for (int i = 0; i < 2; ++i)
#pragma unroll
      for (int j = 0; j < 4; ++j) c[i][j] = wmma_bf16(a[i], b[j], c[i][j]);
  }

  // C layout: element e -> row (m0+16i)+e+8*hl, col n0+16j+l16
#pragma unroll
  for (int i = 0; i < 2; ++i)
#pragma unroll
    for (int j = 0; j < 4; ++j)
#pragma unroll
      for (int e = 0; e < 8; ++e) {
        unsigned off = (unsigned)(m0 + 16 * i + e + 8 * hl) * (unsigned)ldc +
                       (unsigned)(n0 + 16 * j + l16);
        C[off] = c[i][j][e];
      }
}

// ---------------------------------------------------------------------------
// 3) RoPE (neox rotate-half), fp32 math; optional fp32 out + bf16 out.
// ---------------------------------------------------------------------------
__global__ void k_rope(const float* __restrict__ in, const int* __restrict__ pos,
                       float* __restrict__ outF, bf16* __restrict__ outB,
                       int nHeads) {
  int idx = blockIdx.x * blockDim.x + threadIdx.x;
  int total = T_SEQ * nHeads * (HD / 2);
  if (idx >= total) return;
  int d = idx & 63;
  int h = (idx >> 6) % nHeads;
  int t = idx / (64 * nHeads);
  unsigned row = (unsigned)t * (unsigned)(nHeads * HD) + (unsigned)h * HD;
  float x1 = in[row + d];
  float x2 = in[row + d + 64];
  float p = (float)pos[t];
  // inv_freq = 500000^(-2d/128); ln(500000) = 13.122363377404329
  float inv = __expf(-((float)(2 * d) / (float)HD) * 13.1223633774f);
  float ang = p * inv;
  float sn, cs;
  __sincosf(ang, &sn, &cs);
  float o1 = x1 * cs - x2 * sn;
  float o2 = x2 * cs + x1 * sn;
  if (outF) {
    outF[row + d] = o1;
    outF[row + d + 64] = o2;
  }
  outB[row + d] = (bf16)o1;
  outB[row + d + 64] = (bf16)o2;
}

// ---------------------------------------------------------------------------
// 4) V transpose: vF[t][c] (c = hkv*128+d) -> vT[c][t] bf16
// ---------------------------------------------------------------------------
__global__ void k_vt(const float* __restrict__ vF, bf16* __restrict__ vT) {
  int idx = blockIdx.x * blockDim.x + threadIdx.x;
  if (idx >= T_SEQ * NKV * HD) return;
  unsigned c = (unsigned)idx & (NKV * HD - 1);
  unsigned t = (unsigned)idx >> 10;  // NKV*HD = 1024
  vT[c * T_SEQ + t] = (bf16)vF[idx];
}

// ---------------------------------------------------------------------------
// 5) Flash attention: wave = one head x 16 q-rows; block = 8 waves = 128-row
//    q slab of one head. Online softmax, causal, P staged via LDS.
// ---------------------------------------------------------------------------
__global__ __launch_bounds__(256)
void k_attn(const bf16* __restrict__ Qb, const bf16* __restrict__ Kb,
            const bf16* __restrict__ Vt, bf16* __restrict__ Ob) {
  __shared__ bf16 ldsP[8][16][32];  // per-wave P staging tile (8 KB)

  const int lane = threadIdx.x & 31;
  const int wave = threadIdx.x >> 5;
  const int l16  = lane & 15;
  const int hl   = lane >> 4;
  const int h    = blockIdx.x >> 4;                      // q head
  const int qb   = (blockIdx.x & 15) * 128 + wave * 16;  // q-row base
  const int hkv  = h >> 2;                               // GQA: h // 4

  // Q fragments: 16 rows x 128 = 4 A-frags of 16x32
  v16bf qf[4];
  {
    unsigned q0 = (unsigned)(qb + l16) * HID + (unsigned)h * HD + (hl ? 16u : 0u);
#pragma unroll
    for (int j = 0; j < 4; ++j) qf[j] = load_a_frag(Qb, q0 + 32u * j);
  }

  v8f o[8];
#pragma unroll
  for (int j = 0; j < 8; ++j) o[j] = zero8();
  float mrow[8], lrow[8];
#pragma unroll
  for (int e = 0; e < 8; ++e) {
    mrow[e] = -3.0e38f;
    lrow[e] = 0.f;
  }

  const float scale = 0.0883883476483f;  // 1/sqrt(128)
  const int kvEnd = qb + 16;             // causal bound of this tile
  const unsigned kBase = (unsigned)hkv * HD + (hl ? 16u : 0u);
  const unsigned vBase = ((unsigned)hkv * HD + l16) * T_SEQ + (hl ? 16u : 0u);

  for (int kv0 = 0; kv0 < kvEnd; kv0 += 32) {
    // ---- S = Q K^T for 32 keys (two 16x16 C-frags) ----
    v8f s0 = zero8(), s1 = zero8();
    unsigned k0 = (unsigned)(kv0 + l16) * (NKV * HD) + kBase;
    unsigned k1 = (unsigned)(kv0 + 16 + l16) * (NKV * HD) + kBase;
#pragma unroll
    for (int j = 0; j < 4; ++j) {
      s0 = wmma_bf16(qf[j], load_b_frag_bf16(Kb, k0 + 32u * j), s0);
      s1 = wmma_bf16(qf[j], load_b_frag_bf16(Kb, k1 + 32u * j), s1);
    }

    // ---- online softmax (row stats per-lane per C element) ----
    const int col0 = kv0 + l16;
    const int col1 = kv0 + 16 + l16;
    float alpha[8];
#pragma unroll
    for (int e = 0; e < 8; ++e) {
      int row = qb + e + 8 * hl;
      float v0 = (col0 <= row) ? s0[e] * scale : -3.0e38f;
      float v1 = (col1 <= row) ? s1[e] * scale : -3.0e38f;
      float cm = fmaxf(v0, v1);
#pragma unroll
      for (int x = 1; x < 16; x <<= 1) cm = fmaxf(cm, __shfl_xor(cm, x, 32));
      float mn = fmaxf(mrow[e], cm);
      float al = __expf(mrow[e] - mn);
      float p0 = __expf(v0 - mn);
      float p1 = __expf(v1 - mn);
      float rs = p0 + p1;
#pragma unroll
      for (int x = 1; x < 16; x <<= 1) rs += __shfl_xor(rs, x, 32);
      lrow[e] = lrow[e] * al + rs;
      mrow[e] = mn;
      alpha[e] = al;
      // stage P (bf16) row-major 16x32 in this wave's LDS slice
      ldsP[wave][e + 8 * hl][l16] = (bf16)p0;
      ldsP[wave][e + 8 * hl][16 + l16] = (bf16)p1;
    }
    asm volatile("s_wait_dscnt 0" ::: "memory");

    // reload P as A-fragment (16x32)
    Frag16 up;
    {
      const bf16* pp = &ldsP[wave][l16][hl ? 16 : 0];
      up.h[0] = *(const v8bf*)pp;
      up.h[1] = *(const v8bf*)(pp + 8);
    }

    // ---- O = O*alpha + P @ V (8 d-frags of 16 cols) ----
#pragma unroll
    for (int j = 0; j < 8; ++j) {
      v8f oo = o[j];
#pragma unroll
      for (int e = 0; e < 8; ++e) oo[e] *= alpha[e];
      v16bf bv = load_b_frag_bf16(Vt, vBase + (unsigned)(j * 16) * T_SEQ + kv0);
      o[j] = wmma_bf16(up.v, bv, oo);
    }
  }

  // ---- normalize, write bf16 attention output [T, NH*HD] ----
  unsigned oBase = (unsigned)h * HD + (unsigned)l16;
#pragma unroll
  for (int j = 0; j < 8; ++j) {
#pragma unroll
    for (int e = 0; e < 8; ++e) {
      unsigned row = (unsigned)(qb + e + 8 * hl);
      float val = o[j][e] / lrow[e];
      Ob[row * HID + oBase + 16u * j] = (bf16)val;
    }
  }
}

// ---------------------------------------------------------------------------
// Launcher
// ---------------------------------------------------------------------------
extern "C" void kernel_launch(void* const* d_in, const int* in_sizes, int n_in,
                              void* d_out, int out_size, void* d_ws,
                              size_t ws_size, hipStream_t stream) {
  const int*   positions = (const int*)d_in[0];
  const float* hs = (const float*)d_in[1];
  const float* Wq = (const float*)d_in[2];
  const float* Wk = (const float*)d_in[3];
  const float* Wv = (const float*)d_in[4];
  const float* Wo = (const float*)d_in[5];

  float* out  = (float*)d_out;                          // [2048, 4096]
  float* kOut = out + (size_t)T_SEQ * HID;              // [2048, 1024] (roped)
  float* vOut = kOut + (size_t)T_SEQ * NKV * HD;        // [2048, 1024]

  // workspace layout (~80 MB)
  char* ws = (char*)d_ws;
  bf16*  hsB  = (bf16*)ws;  ws += (size_t)T_SEQ * HID * 2;        // 16 MB
  float* qRaw = (float*)ws;                                       // 32 MB
  bf16*  attnB = (bf16*)qRaw;  // reuses qRaw region after RoPE consumes it
  ws += (size_t)T_SEQ * HID * 4;
  float* kRaw = (float*)ws; ws += (size_t)T_SEQ * NKV * HD * 4;   //  8 MB
  bf16*  qB   = (bf16*)ws;  ws += (size_t)T_SEQ * HID * 2;        // 16 MB
  bf16*  kB   = (bf16*)ws;  ws += (size_t)T_SEQ * NKV * HD * 2;   //  4 MB
  bf16*  vT   = (bf16*)ws;  ws += (size_t)T_SEQ * NKV * HD * 2;   //  4 MB

  // 1) hidden -> bf16
  {
    int n = T_SEQ * HID;
    k_cvt_bf16<<<(n + 255) / 256, 256, 0, stream>>>(hs, hsB, n);
  }

  // 2) projections (weights converted to bf16 on the fly; L2-resident)
  k_gemm_bf16w32<<<dim3(HID / 128, T_SEQ / 128), 256, 0, stream>>>(
      hsB, Wq, qRaw, HID);
  k_gemm_bf16w32<<<dim3((NKV * HD) / 128, T_SEQ / 128), 256, 0, stream>>>(
      hsB, Wk, kRaw, NKV * HD);
  k_gemm_bf16w32<<<dim3((NKV * HD) / 128, T_SEQ / 128), 256, 0, stream>>>(
      hsB, Wv, vOut, NKV * HD);

  // 3) RoPE: Q -> bf16 only; K -> fp32 output + bf16
  {
    int n = T_SEQ * NH * (HD / 2);
    k_rope<<<(n + 255) / 256, 256, 0, stream>>>(qRaw, positions, nullptr, qB, NH);
  }
  {
    int n = T_SEQ * NKV * (HD / 2);
    k_rope<<<(n + 255) / 256, 256, 0, stream>>>(kRaw, positions, kOut, kB, NKV);
  }

  // 4) V transpose to [d][t] bf16 for contiguous P@V B-fragments
  {
    int n = T_SEQ * NKV * HD;
    k_vt<<<(n + 255) / 256, 256, 0, stream>>>(vOut, vT);
  }

  // 5) flash attention -> bf16 [T, NH*HD] (into reused qRaw region)
  k_attn<<<dim3(NH * (T_SEQ / 128)), 256, 0, stream>>>(qB, kB, vT, attnB);

  // 6) output projection
  k_gemm_bf16w32<<<dim3(HID / 128, T_SEQ / 128), 256, 0, stream>>>(
      attnB, Wo, out, HID);
}
